// AttentionBlock_26371099197899
// MI455X (gfx1250) — matrix operand order repaired and checked by reading, not compiled, not run on hardware
//
#include <hip/hip_runtime.h>
#include <hip/hip_bf16.h>

typedef _Float16 half_t;
typedef __attribute__((ext_vector_type(16))) _Float16 v16h;
typedef __attribute__((ext_vector_type(8)))  _Float16 v8h;
typedef __attribute__((ext_vector_type(8)))  float    v8f;
typedef __attribute__((ext_vector_type(4)))  int      v4i;

#define WMMA_F32_F16(a, b, c) \
  __builtin_amdgcn_wmma_f32_16x16x32_f16(false, (a), false, (b), (short)0, (c), false, false)

// --- CDNA5 async global->LDS copy (ASYNCcnt-tracked), with guarded fallback --
// Probe-confirmed: builtin exists; first param is int4 in AS(1) (prints as
// "int __vector(4) __device__ *"), so pass (v4i AS1*, v4i AS3*, imm, imm).
__device__ __forceinline__ void async_b128(const half_t* g, half_t* l) {
#if __has_builtin(__builtin_amdgcn_global_load_async_to_lds_b128)
  __builtin_amdgcn_global_load_async_to_lds_b128(
      (__attribute__((address_space(1))) v4i*)g,
      (__attribute__((address_space(3))) v4i*)l, 0, 0);
#else
  *(v8h*)l = *(const v8h*)g;
#endif
}

#if __has_builtin(__builtin_amdgcn_s_wait_asynccnt)
#define WAIT_ASYNC(n) __builtin_amdgcn_s_wait_asynccnt(n)
#else
#if __has_builtin(__builtin_amdgcn_global_load_async_to_lds_b128)
#define WAIT_ASYNC(n) asm volatile("s_wait_asynccnt %0" ::"i"(n) : "memory")
#else
#define WAIT_ASYNC(n) ((void)0)
#endif
#endif

// ---------------------------------------------------------------------------
// Fragment loaders. Source tiles are row-major [outer][K] with `stride` in
// halfs (multiple of 8 so v8h loads stay 16B aligned).
//
// A (16x32 f16): lane L -> row = L&15; K halves {k0..k0+7, k0+16..k0+23},
//                k0 = (L<16 ? 0 : 8).   (ISA 7.12.2, 16-bit A 16x32)
// B (32x16 f16): lane L -> col = L&15; K halves {k0..k0+7, k0+16..k0+23},
//                k0 = (L<16 ? 0 : 8).   (row-per-output-column "BT" layout)
// C/D (16x16 f32): VGPR r -> M = r + (L<16?0:8), N = L&15.
// ---------------------------------------------------------------------------
template <typename P>
__device__ __forceinline__ v16h load_frag_a(P base, int stride, int lane) {
  const int row = lane & 15;
  const int k0  = (lane < 16) ? 0 : 8;
  v8h lo = *(const v8h*)(&base[row * stride + k0]);
  v8h hi = *(const v8h*)(&base[row * stride + k0 + 16]);
  v16h r;
#pragma unroll
  for (int i = 0; i < 8; ++i) { r[i] = lo[i]; r[i + 8] = hi[i]; }
  return r;
}

template <typename P>
__device__ __forceinline__ v16h load_frag_b(P base, int stride, int lane) {
  const int col = lane & 15;
  const int k0  = (lane < 16) ? 0 : 8;
  v8h lo = *(const v8h*)(&base[col * stride + k0]);
  v8h hi = *(const v8h*)(&base[col * stride + k0 + 16]);
  v16h r;
#pragma unroll
  for (int i = 0; i < 8; ++i) { r[i] = lo[i]; r[i + 8] = hi[i]; }
  return r;
}

// ---------------------------------------------------------------------------
// Kernel 1: fused QKV projection.
//   xt[b][n][c] = x[b][c][n];  q/k = xt@W + bias  -> f16 (b,n,c)
//   v stored transposed: vT[b][c][n]              -> f16 (b,c,n)
// ---------------------------------------------------------------------------
__global__ __launch_bounds__(256) void qkv_kernel(
    const float* __restrict__ x,
    const float* __restrict__ qw, const float* __restrict__ qb,
    const float* __restrict__ kw, const float* __restrict__ kb,
    const float* __restrict__ vw, const float* __restrict__ vb,
    half_t* __restrict__ qo, half_t* __restrict__ ko, half_t* __restrict__ vto) {
  __shared__ half_t xs[64 * 136];    // [n_local][c], stride 136 halfs
  __shared__ half_t wst[128 * 136];  // [j][c] = W^T, stride 136 halfs

  const int tid  = threadIdx.x;
  const int wave = tid >> 5;
  const int lane = tid & 31;
  const int b    = blockIdx.y;
  const int n0   = blockIdx.x * 64;

  for (int i = tid; i < 64 * 128; i += 256) {
    const int c  = i >> 6;
    const int nl = i & 63;
    xs[nl * 136 + c] = (half_t)x[((size_t)(b * 128 + c)) * 4096 + n0 + nl];
  }

  const int rt      = wave & 3;
  const int colbase = (wave >> 2) * 64;
  const int r0      = rt * 16;
  const int rbase   = (lane < 16) ? 0 : 8;

  for (int widx = 0; widx < 3; ++widx) {
    const float* W    = (widx == 0) ? qw : ((widx == 1) ? kw : vw);
    const float* bias = (widx == 0) ? qb : ((widx == 1) ? kb : vb);

    __syncthreads();
    for (int i = tid; i < 128 * 128; i += 256) {
      const int c = i >> 7;
      const int j = i & 127;
      wst[j * 136 + c] = (half_t)W[c * 128 + j];
    }
    __syncthreads();

#pragma unroll
    for (int ct = 0; ct < 4; ++ct) {
      const int col0 = colbase + ct * 16;
      v8f acc = {};
#pragma unroll
      for (int kc = 0; kc < 4; ++kc) {
        v16h a  = load_frag_a(&xs[r0 * 136 + kc * 32], 136, lane);
        v16h bm = load_frag_b(&wst[col0 * 136 + kc * 32], 136, lane);
        acc = WMMA_F32_F16(a, bm, acc);
      }
      const int cc  = col0 + (lane & 15);
      const float bb = bias[cc];
#pragma unroll
      for (int r = 0; r < 8; ++r) {
        const int mrow = n0 + r0 + rbase + r;
        const float val = acc[r] + bb;
        if (widx == 2) {
          vto[((size_t)(b * 128 + cc)) * 4096 + mrow] = (half_t)val;
        } else {
          half_t* dst = (widx == 0) ? qo : ko;
          dst[((size_t)(b * 4096 + mrow)) * 128 + cc] = (half_t)val;
        }
      }
    }
  }
}

// ---------------------------------------------------------------------------
// Kernel 2: flash attention, async double-buffered K/V staging.
// Block: 256 thr = 8 waves; each wave owns 16 query rows (block = 128 rows).
// Per 32-key chunk: async engine stages K (32x128) and V^T (128x32) into LDS
// once per block while WMMAs consume the previous chunk.
// ---------------------------------------------------------------------------
#define KSTRIDE 136  // halfs; 32 rows (keys) x 128 channels
#define VSTRIDE 40   // halfs; 128 rows (channels) x 32 keys

__device__ __forceinline__ void prefetch_chunk(const half_t* kg, const half_t* vg,
                                               half_t* kbuf, half_t* vbuf, int tid) {
  // K chunk: 32 rows x 128 halfs; 8 threads/row, 32B each.
  {
    const int row = tid >> 3;
    const int c0  = (tid & 7) * 16;
    const half_t* g = kg + row * 128 + c0;
    half_t*       l = kbuf + row * KSTRIDE + c0;
    async_b128(g, l);
    async_b128(g + 8, l + 8);
  }
  // V chunk (from vT): 128 rows x 32 halfs; 2 threads/row, 32B each.
  {
    const int row = tid >> 1;
    const int c0  = (tid & 1) * 16;
    const half_t* g = vg + (size_t)row * 4096 + c0;
    half_t*       l = vbuf + row * VSTRIDE + c0;
    async_b128(g, l);
    async_b128(g + 8, l + 8);
  }
}

__global__ __launch_bounds__(256) void attn_kernel(
    const half_t* __restrict__ q, const half_t* __restrict__ kmat,
    const half_t* __restrict__ vt, half_t* __restrict__ ho) {
  __shared__ half_t ksh[2][32 * KSTRIDE];
  __shared__ half_t vsh[2][128 * VSTRIDE];
  __shared__ half_t ps[8 * 16 * 40];  // per-wave 16x32 P tile

  const int tid   = threadIdx.x;
  const int wave  = tid >> 5;
  const int lane  = tid & 31;
  const int b     = blockIdx.y;
  const int row0  = blockIdx.x * 128 + wave * 16;
  const int col   = lane & 15;
  const int rbase = (lane < 16) ? 0 : 8;
  const float scale = 0.08838834764831845f;  // 128^-0.5

  const half_t* kbase = kmat + (size_t)b * 4096 * 128;
  const half_t* vbase = vt + (size_t)b * 128 * 4096;

  const half_t* qp = q + ((size_t)b * 4096 + row0) * 128;
  v16h qa[4];
#pragma unroll
  for (int i = 0; i < 4; ++i) qa[i] = load_frag_a(qp + i * 32, 128, lane);

  v8f o[8] = {};
  float m[8], l[8];
#pragma unroll
  for (int r = 0; r < 8; ++r) { m[r] = -3.0e38f; l[r] = 0.0f; }

  half_t* pw = &ps[wave * 16 * 40];

  int buf = 0;
  prefetch_chunk(kbase, vbase, ksh[0], vsh[0], tid);

  for (int kb = 0; kb < 4096; kb += 32) {
    if (kb + 32 < 4096) {
      prefetch_chunk(kbase + (size_t)(kb + 32) * 128, vbase + (kb + 32),
                     ksh[buf ^ 1], vsh[buf ^ 1], tid);
      WAIT_ASYNC(4);  // async loads complete in order: current chunk landed
    } else {
      WAIT_ASYNC(0);
    }
    __syncthreads();  // current chunk visible to all waves

    const half_t* kt = ksh[buf];
    const half_t* vtile = vsh[buf];

    v8f s0 = {}, s1 = {};
#pragma unroll
    for (int c2 = 0; c2 < 4; ++c2) {
      v16h b0 = load_frag_b(kt + c2 * 32, KSTRIDE, lane);                // keys 0..15
      s0 = WMMA_F32_F16(qa[c2], b0, s0);
      v16h b1 = load_frag_b(kt + 16 * KSTRIDE + c2 * 32, KSTRIDE, lane); // keys 16..31
      s1 = WMMA_F32_F16(qa[c2], b1, s1);
    }

    float p0v[8], p1v[8], alpha[8];
#pragma unroll
    for (int r = 0; r < 8; ++r) {
      const float a0 = s0[r] * scale;
      const float a1 = s1[r] * scale;
      float mv = fmaxf(a0, a1);  // row reduce within 16-lane halves
      mv = fmaxf(mv, __shfl_xor(mv, 1));
      mv = fmaxf(mv, __shfl_xor(mv, 2));
      mv = fmaxf(mv, __shfl_xor(mv, 4));
      mv = fmaxf(mv, __shfl_xor(mv, 8));
      const float mn = fmaxf(m[r], mv);
      alpha[r] = __expf(m[r] - mn);
      const float e0 = __expf(a0 - mn);
      const float e1 = __expf(a1 - mn);
      float rs = e0 + e1;
      rs += __shfl_xor(rs, 1);
      rs += __shfl_xor(rs, 2);
      rs += __shfl_xor(rs, 4);
      rs += __shfl_xor(rs, 8);
      l[r] = l[r] * alpha[r] + rs;
      m[r] = mn;
      p0v[r] = e0;
      p1v[r] = e1;
    }

#pragma unroll
    for (int j = 0; j < 8; ++j)
#pragma unroll
      for (int r = 0; r < 8; ++r) o[j][r] *= alpha[r];

    // D-layout -> A-layout via per-wave LDS region (same-wave DS stays in order).
#pragma unroll
    for (int r = 0; r < 8; ++r) {
      pw[(rbase + r) * 40 + col]      = (half_t)p0v[r];
      pw[(rbase + r) * 40 + col + 16] = (half_t)p1v[r];
    }
    v16h pa = load_frag_a(pw, 40, lane);

#pragma unroll
    for (int j = 0; j < 8; ++j) {
      v16h vb_ = load_frag_b(vtile + (j * 16) * VSTRIDE, VSTRIDE, lane);
      o[j] = WMMA_F32_F16(pa, vb_, o[j]);
    }

    __syncthreads();  // all waves done reading buf before async overwrites it
    buf ^= 1;
  }

  float linv[8];
#pragma unroll
  for (int r = 0; r < 8; ++r) linv[r] = 1.0f / l[r];
#pragma unroll
  for (int j = 0; j < 8; ++j)
#pragma unroll
    for (int r = 0; r < 8; ++r) {
      const int mrow = row0 + rbase + r;
      ho[((size_t)b * 4096 + mrow) * 128 + j * 16 + col] = (half_t)(o[j][r] * linv[r]);
    }
}

// ---------------------------------------------------------------------------
// Kernel 3: out = xt + hout @ proj_w + proj_b   (fp32 output, (b,n,c) layout)
// ---------------------------------------------------------------------------
__global__ __launch_bounds__(256) void proj_kernel(
    const half_t* __restrict__ ho, const float* __restrict__ pwght,
    const float* __restrict__ pb, const float* __restrict__ x,
    float* __restrict__ out) {
  __shared__ half_t wst[128 * 136];  // proj_w^T [j][c]

  const int tid  = threadIdx.x;
  const int wave = tid >> 5;
  const int lane = tid & 31;
  const int b    = blockIdx.y;
  const int n0   = blockIdx.x * 64;

  for (int i = tid; i < 128 * 128; i += 256) {
    const int c = i >> 7;
    const int j = i & 127;
    wst[j * 136 + c] = (half_t)pwght[c * 128 + j];
  }
  __syncthreads();

  const int rt      = wave & 3;
  const int colbase = (wave >> 2) * 64;
  const int r0      = rt * 16;
  const int rbase   = (lane < 16) ? 0 : 8;

  const half_t* ap = ho + ((size_t)b * 4096 + n0 + r0) * 128;
  v16h a[4];
#pragma unroll
  for (int i = 0; i < 4; ++i) a[i] = load_frag_a(ap + i * 32, 128, lane);

#pragma unroll
  for (int ct = 0; ct < 4; ++ct) {
    const int col0 = colbase + ct * 16;
    v8f acc = {};
#pragma unroll
    for (int kc = 0; kc < 4; ++kc) {
      v16h bm = load_frag_b(&wst[col0 * 136 + kc * 32], 136, lane);
      acc = WMMA_F32_F16(a[kc], bm, acc);
    }
    const int cc  = col0 + (lane & 15);
    const float bb = pb[cc];
#pragma unroll
    for (int r = 0; r < 8; ++r) {
      const int mrow = n0 + r0 + rbase + r;
      const float resid = x[((size_t)(b * 128 + cc)) * 4096 + mrow];
      out[((size_t)(b * 4096 + mrow)) * 128 + cc] = acc[r] + bb + resid;
    }
  }
}

// ---------------------------------------------------------------------------
extern "C" void kernel_launch(void* const* d_in, const int* in_sizes, int n_in,
                              void* d_out, int out_size, void* d_ws, size_t ws_size,
                              hipStream_t stream) {
  const float* x  = (const float*)d_in[0];
  const float* qw = (const float*)d_in[1];
  const float* qb = (const float*)d_in[2];
  const float* kw = (const float*)d_in[3];
  const float* kb = (const float*)d_in[4];
  const float* vw = (const float*)d_in[5];
  const float* vb = (const float*)d_in[6];
  const float* pw = (const float*)d_in[7];
  const float* pb = (const float*)d_in[8];
  float* out = (float*)d_out;

  const size_t per = (size_t)8 * 4096 * 128;  // elements per f16 tensor
  half_t* qws  = (half_t*)d_ws;
  half_t* kws  = qws + per;
  half_t* vtws = kws + per;
  half_t* hows = vtws + per;  // total 32 MB of workspace

  qkv_kernel<<<dim3(64, 8), 256, 0, stream>>>(x, qw, qb, kw, kb, vw, vb,
                                              qws, kws, vtws);
  attn_kernel<<<dim3(32, 8), 256, 0, stream>>>(qws, kws, vtws, hows);
  proj_kernel<<<dim3(64, 8), 256, 0, stream>>>(hows, pw, pb, x, out);
}
